// seqFusionAttentionNoBiasDiff_33002528703224
// MI455X (gfx1250) — matrix operand order repaired and checked by reading, not compiled, not run on hardware
//
#include <hip/hip_runtime.h>

#define HNUM 16
#define CDIM 64

typedef __attribute__((ext_vector_type(16))) __bf16 v16bf;
typedef __attribute__((ext_vector_type(8)))  __bf16 v8bf;
typedef __attribute__((ext_vector_type(8)))  float  v8f;
typedef __attribute__((ext_vector_type(4)))  float  v4f;
typedef __attribute__((ext_vector_type(4)))  unsigned v4u;
typedef __attribute__((ext_vector_type(8)))  unsigned v8u;

union BF16x16 { v16bf v; __bf16 e[16]; };
union BF16x8  { v8bf  v; __bf16 e[8];  };

__device__ __forceinline__ __bf16 f2bf(float f) {
  unsigned u = __builtin_bit_cast(unsigned, f);
  u += 0x7FFFu + ((u >> 16) & 1u);            // round-to-nearest-even
  unsigned short h = (unsigned short)(u >> 16);
  return __builtin_bit_cast(__bf16, h);
}

// Packed f32x2 -> bf16x2 conversion: prefer v_cvt_pk_bf16_f32 when available.
#if __has_builtin(__builtin_amdgcn_cvt_pk_bf16_f32)
#define CVT_PK_BF16(dstptr, x, y)                                   \
  do {                                                              \
    auto _p = __builtin_amdgcn_cvt_pk_bf16_f32((x), (y));           \
    __builtin_memcpy((void*)(dstptr), &_p, 4);                      \
  } while (0)
#else
#define CVT_PK_BF16(dstptr, x, y)                                   \
  do {                                                              \
    __bf16* _d = (__bf16*)(dstptr);                                 \
    _d[0] = f2bf(x); _d[1] = f2bf(y);                               \
  } while (0)
#endif

// ---- CDNA5 async global->LDS copy (ASYNCcnt) ------------------------------
__device__ __forceinline__ void async_copy_b128(unsigned lds_off,
                                                unsigned long long gaddr) {
  asm volatile("global_load_async_to_lds_b128 %0, %1, off"
               :: "v"(lds_off), "v"(gaddr) : "memory");
}
template <short N> __device__ __forceinline__ void wait_async() {
#if __has_builtin(__builtin_amdgcn_s_wait_asynccnt)
  __builtin_amdgcn_s_wait_asynccnt(N);
#else
  asm volatile("s_wait_asynccnt %0" :: "i"(N) : "memory");
#endif
}

// ---- CDNA5 Tensor Data Mover: 2D bf16 tile load (TENSORcnt) ---------------
// D# group0: count=1 | lds_addr | global_addr | type=2
// D# group1: data_size=2B, tensor_dim0/1, tile_dim0/1, tensor_dim0_stride
__device__ __forceinline__ void tdm_load_tile_2d(unsigned lds_off,
                                                 unsigned long long gaddr,
                                                 unsigned tensor_d0,
                                                 unsigned tensor_d1,
                                                 unsigned tile_d0,
                                                 unsigned tile_d1,
                                                 unsigned stride_d0) {
  v4u g0;
  g0[0] = 1u;                                           // count = 1
  g0[1] = lds_off;                                      // lds_addr
  g0[2] = (unsigned)gaddr;                              // global_addr[31:0]
  g0[3] = ((unsigned)(gaddr >> 32) & 0x01FFFFFFu)       // global_addr[56:32]
          | (2u << 30);                                 // type = 2 ("image")
  v8u g1;
  g1[0] = 1u << 16;                                     // data_size = 2 bytes
  g1[1] = (tensor_d0 & 0xFFFFu) << 16;                  // tensor_dim0[15:0]
  g1[2] = (tensor_d0 >> 16) | ((tensor_d1 & 0xFFFFu) << 16);
  g1[3] = (tensor_d1 >> 16) | (tile_d0 << 16);          // tile_dim0
  g1[4] = tile_d1;                                      // tile_dim1 (dim2 = 0)
  g1[5] = stride_d0;                                    // tensor_dim0_stride
  g1[6] = 0;
  g1[7] = 0;
  v4u z4 = {0u, 0u, 0u, 0u};
  asm volatile("tensor_load_to_lds %0, %1, %2, %3"
               :: "s"(g0), "s"(g1), "s"(z4), "s"(z4) : "memory");
}
template <short N> __device__ __forceinline__ void wait_tensor() {
#if __has_builtin(__builtin_amdgcn_s_wait_tensorcnt)
  __builtin_amdgcn_s_wait_tensorcnt(N);
#else
  asm volatile("s_wait_tensorcnt %0" :: "i"(N) : "memory");
#endif
}

// ---------------------------------------------------------------------------
// Generic GEMM: C[M,N] = A[M,K] x B[K,N], fp32 in memory, bf16 WMMA compute.
// WG = 256 threads (8 waves), tile 64x64, K-step 64 (2 wmma k-steps / stage).
// mode 0: fp32 out (+bias);  mode 1: bf16 out packed as [B, H, S, C].
// ---------------------------------------------------------------------------
__global__ __launch_bounds__(256)
void gemm_bf16_wmma(const float* __restrict__ A, const float* __restrict__ Bm,
                    const float* __restrict__ bias,
                    float* __restrict__ outF, __bf16* __restrict__ outBF,
                    int M, int N, int K, int Sseq, int mode)
{
  __shared__ __align__(16) __bf16 As[64][64];   // [m][k]
  __shared__ __align__(16) __bf16 BsT[64][64];  // [n][k] (transposed)

  const int t    = threadIdx.x;
  const int lane = t & 31;
  const int wave = t >> 5;
  const int half = lane >> 4;
  const int l16  = lane & 15;
  const int wm   = wave >> 2;          // 0..1
  const int wn   = wave & 3;           // 0..3
  const long blockN = (long)blockIdx.x * 64;
  const long blockM = (long)blockIdx.y * 64;

  v8f acc[2] = {};

  for (int k0 = 0; k0 < K; k0 += 64) {
    #pragma unroll
    for (int i = 0; i < 4; ++i) {
      int idx = t + i * 256;
      int r = idx >> 4, c4 = (idx & 15) << 2;
      v4f a = *(const v4f*)(A + (blockM + r) * (long)K + k0 + c4);
      CVT_PK_BF16(&As[r][c4], a.x, a.y);
      CVT_PK_BF16(&As[r][c4 + 2], a.z, a.w);
    }
    #pragma unroll
    for (int i = 0; i < 4; ++i) {
      int idx = t + i * 256;
      int r = idx >> 4, c4 = (idx & 15) << 2;
      v4f bv = *(const v4f*)(Bm + (long)(k0 + r) * N + blockN + c4);
      BsT[c4 + 0][r] = f2bf(bv.x);
      BsT[c4 + 1][r] = f2bf(bv.y);
      BsT[c4 + 2][r] = f2bf(bv.z);
      BsT[c4 + 3][r] = f2bf(bv.w);
    }
    if (k0 + 64 < K) {
      __builtin_prefetch(A + (blockM + (t >> 2)) * (long)K + (k0 + 64), 0, 0);
      __builtin_prefetch(Bm + (long)(k0 + 64 + (t >> 3)) * N + blockN, 0, 0);
    }
    __syncthreads();

    #pragma unroll
    for (int ks = 0; ks < 2; ++ks) {
      BF16x16 bfr;
      bfr.v = *(const v16bf*)&BsT[wn * 16 + l16][ks * 32 + half * 16];
      #pragma unroll
      for (int ti = 0; ti < 2; ++ti) {
        const __bf16* ar = &As[(2 * wm + ti) * 16 + l16][ks * 32];
        BF16x8 lo, hi;
        lo.v = *(const v8bf*)(ar + half * 16);
        hi.v = *(const v8bf*)(ar + half * 16 + 8);
        BF16x16 afr;
        #pragma unroll
        for (int j = 0; j < 8; ++j) { afr.e[j] = lo.e[j]; afr.e[8 + j] = hi.e[j]; }
        acc[ti] = __builtin_amdgcn_wmma_f32_16x16x32_bf16(
            false, afr.v, false, bfr.v, (short)0, acc[ti], false, false);
      }
    }
    __syncthreads();
  }

  const int col  = (int)blockN + wn * 16 + l16;
  const float bcol = (mode == 0 && bias) ? bias[col] : 0.0f;
  #pragma unroll
  for (int ti = 0; ti < 2; ++ti) {
    #pragma unroll
    for (int r = 0; r < 8; ++r) {
      int row = (int)blockM + (2 * wm + ti) * 16 + r + 8 * half;
      float v = acc[ti][r];
      if (mode == 0) {
        outF[(long)row * N + col] = v + bcol;
      } else {
        int b_ = row / Sseq, s_ = row % Sseq;
        int h_ = col >> 6, c_ = col & 63;
        outBF[(((long)(b_ * HNUM + h_) * Sseq + s_) << 6) + c_] = f2bf(v);
      }
    }
  }
}

// ---------------------------------------------------------------------------
// VG = V * sigmoid(G), packed bf16 [B, H, S, C]
// ---------------------------------------------------------------------------
__global__ __launch_bounds__(256)
void pack_vg(const float* __restrict__ Vp, const float* __restrict__ Gp,
             __bf16* __restrict__ VGh, int Sseq, int total)
{
  int i = blockIdx.x * 256 + threadIdx.x;
  if (i >= total) return;
  int m = i >> 10;            // N = 1024
  int n = i & 1023;
  float g = Gp[i];
  float v = Vp[i] / (1.0f + __expf(-g));
  int b_ = m / Sseq, s_ = m % Sseq;
  int h_ = n >> 6, c_ = n & 63;
  VGh[(((long)(b_ * HNUM + h_) * Sseq + s_) << 6) + c_] = f2bf(v);
}

// ---------------------------------------------------------------------------
// Flash attention: one WG per (b, h, 64-query block). 8 waves.
// K blocks double-buffered via global_load_async_to_lds_b128 (ASYNCcnt),
// VG blocks double-buffered via the Tensor Data Mover (TENSORcnt, wave 0).
// S = Q K^T and P @ VG both on bf16 WMMA, fp32 accumulate.
// ---------------------------------------------------------------------------
__global__ __launch_bounds__(256)
void attn_fwd(const __bf16* __restrict__ Qh, const __bf16* __restrict__ Kh,
              const __bf16* __restrict__ VGh, float* __restrict__ ctx, int Sseq)
{
  __shared__ __align__(16) __bf16 Ks[2][64][64];   // K block, [t][c]
  __shared__ __align__(16) __bf16 VGr[2][64][64];  // VG block (TDM), [t][c]
  __shared__ __align__(16) __bf16 VGsT[64][64];    // VG transposed, [c][t]
  __shared__ __align__(16) float  Sb[64][68];
  __shared__ __align__(16) __bf16 Pb[64][64];
  __shared__ float rowm[64], rowl[64], rowa[64];
  __shared__ float red[64][4];

  const int t    = threadIdx.x;
  const int lane = t & 31;
  const int wave = t >> 5;
  const int half = lane >> 4;
  const int l16  = lane & 15;
  const int mt   = wave >> 1;     // 0..3
  const int nh   = wave & 1;      // 0..1
  const int qb   = blockIdx.x * 64;
  const int h    = blockIdx.y;
  const int b    = blockIdx.z;
  const long headBase = ((long)(b * HNUM + h)) * Sseq * CDIM;

  auto issueK = [&](int jb, int buf) {
    #pragma unroll
    for (int i = 0; i < 2; ++i) {
      int id = t + i * 256;
      int r = id >> 3, cq = (id & 7) * 8;     // 8 bf16 = 16B chunk
      unsigned lds_off = (unsigned)(unsigned long long)(void*)&Ks[buf][r][cq];
      unsigned long long ga = (unsigned long long)
          (Kh + headBase + (long)(jb + r) * CDIM + cq);
      async_copy_b128(lds_off, ga);
    }
  };
  auto issueVG = [&](int jb, int buf) {
    if (wave == 0) {
      unsigned lds_off = (unsigned)(unsigned long long)(void*)&VGr[buf][0][0];
      unsigned long long ga = (unsigned long long)(VGh + headBase + (long)jb * CDIM);
      tdm_load_tile_2d(lds_off, ga, 64u, (unsigned)Sseq, 64u, 64u, 64u);
    }
  };

  // preload Q A-fragments (rows qb + mt*16 + m), 2 k-steps over C=64
  BF16x16 aq[2];
  {
    const __bf16* qrow = Qh + headBase + (long)(qb + mt * 16 + l16) * CDIM;
    #pragma unroll
    for (int ks = 0; ks < 2; ++ks) {
      BF16x8 lo, hi;
      lo.v = *(const v8bf*)(qrow + ks * 32 + half * 16);
      hi.v = *(const v8bf*)(qrow + ks * 32 + half * 16 + 8);
      #pragma unroll
      for (int j = 0; j < 8; ++j) { aq[ks].e[j] = lo.e[j]; aq[ks].e[8 + j] = hi.e[j]; }
    }
  }

  if (t < 64) { rowm[t] = -3.0e38f; rowl[t] = 0.0f; }
  v8f o[2] = {};

  // prologue: stage block 0 into buffer 0
  issueK(0, 0);
  issueVG(0, 0);
  __syncthreads();

  const float scale = 0.125f;  // 1/sqrt(64)
  const int nblk = Sseq / 64;

  for (int ib = 0; ib < nblk; ++ib) {
    const int jb  = ib * 64;
    const int cur = ib & 1;
    const int nxt = cur ^ 1;
    const bool pf = (ib + 1 < nblk);

    // prefetch next block while we consume the current one
    if (pf) { issueK(jb + 64, nxt); issueVG(jb + 64, nxt); }
    if (pf) wait_async<2>(); else wait_async<0>();
    if (wave == 0) { if (pf) wait_tensor<1>(); else wait_tensor<0>(); }
    __syncthreads();

    // ---- transpose VGr[cur] -> VGsT (LDS -> LDS) ----
    {
      const int tr = t >> 2, cq = (t & 3) * 16;
      BF16x16 vv;
      vv.v = *(const v16bf*)&VGr[cur][tr][cq];
      #pragma unroll
      for (int j = 0; j < 16; ++j) VGsT[cq + j][tr] = vv.e[j];
    }
    // ---- S tiles: 2 n-tiles per wave, 2 wmma each (K frags from LDS) ----
    #pragma unroll
    for (int nt2 = 0; nt2 < 2; ++nt2) {
      const int nt = nh * 2 + nt2;
      v8f s = {};
      #pragma unroll
      for (int ks = 0; ks < 2; ++ks) {
        v16bf bk = *(const v16bf*)&Ks[cur][nt * 16 + l16][ks * 32 + half * 16];
        s = __builtin_amdgcn_wmma_f32_16x16x32_bf16(
            false, aq[ks].v, false, bk, (short)0, s, false, false);
      }
      #pragma unroll
      for (int r = 0; r < 8; ++r)
        Sb[mt * 16 + r + 8 * half][nt * 16 + l16] = s[r] * scale;
    }
    __syncthreads();

    // ---- row max (4 threads per row) ----
    {
      const int row = t >> 2, q = t & 3;
      float mx = -3.0e38f;
      #pragma unroll
      for (int j = 0; j < 16; ++j) mx = fmaxf(mx, Sb[row][q * 16 + j]);
      red[row][q] = mx;
    }
    __syncthreads();
    if (t < 64) {
      float mnew = fmaxf(fmaxf(red[t][0], red[t][1]), fmaxf(red[t][2], red[t][3]));
      float mold = rowm[t];
      mnew = fmaxf(mold, mnew);
      rowa[t] = __expf(mold - mnew);
      rowm[t] = mnew;
    }
    __syncthreads();
    // ---- exp, partial sum, P -> bf16 ----
    {
      const int row = t >> 2, q = t & 3;
      const float mnew = rowm[row];
      float sum = 0.0f;
      #pragma unroll
      for (int j = 0; j < 16; j += 2) {
        float p0 = __expf(Sb[row][q * 16 + j + 0] - mnew);
        float p1 = __expf(Sb[row][q * 16 + j + 1] - mnew);
        sum += p0 + p1;
        CVT_PK_BF16(&Pb[row][q * 16 + j], p0, p1);
      }
      red[row][q] = sum;
    }
    __syncthreads();
    if (t < 64)
      rowl[t] = rowl[t] * rowa[t] + (red[t][0] + red[t][1] + red[t][2] + red[t][3]);
    __syncthreads();

    // ---- O rescale + P @ VG ----
    float ar[8];
    #pragma unroll
    for (int r = 0; r < 8; ++r) ar[r] = rowa[mt * 16 + r + 8 * half];
    #pragma unroll
    for (int ct2 = 0; ct2 < 2; ++ct2) {
      #pragma unroll
      for (int r = 0; r < 8; ++r) o[ct2][r] *= ar[r];
    }

    BF16x16 ap[2];
    #pragma unroll
    for (int ks = 0; ks < 2; ++ks) {
      const __bf16* prow = &Pb[mt * 16 + l16][0];
      BF16x8 lo, hi;
      lo.v = *(const v8bf*)(prow + ks * 32 + half * 16);
      hi.v = *(const v8bf*)(prow + ks * 32 + half * 16 + 8);
      #pragma unroll
      for (int j = 0; j < 8; ++j) { ap[ks].e[j] = lo.e[j]; ap[ks].e[8 + j] = hi.e[j]; }
    }
    #pragma unroll
    for (int ct2 = 0; ct2 < 2; ++ct2) {
      const int ct = nh * 2 + ct2;
      #pragma unroll
      for (int ks = 0; ks < 2; ++ks) {
        v16bf bv = *(const v16bf*)&VGsT[ct * 16 + l16][ks * 32 + half * 16];
        o[ct2] = __builtin_amdgcn_wmma_f32_16x16x32_bf16(
            false, ap[ks].v, false, bv, (short)0, o[ct2], false, false);
      }
    }
    __syncthreads();
  }

  // ---- normalize and write ctx as [B, S, H*C] fp32 ----
  float linv[8];
  #pragma unroll
  for (int r = 0; r < 8; ++r) linv[r] = 1.0f / rowl[mt * 16 + r + 8 * half];
  #pragma unroll
  for (int ct2 = 0; ct2 < 2; ++ct2) {
    const int col = (nh * 2 + ct2) * 16 + l16;
    #pragma unroll
    for (int r = 0; r < 8; ++r) {
      const int row = qb + mt * 16 + r + 8 * half;
      ctx[((long)(b * Sseq + row)) * (HNUM * CDIM) + h * CDIM + col] = o[ct2][r] * linv[r];
    }
  }
}

// ---------------------------------------------------------------------------
extern "C" void kernel_launch(void* const* d_in, const int* in_sizes, int n_in,
                              void* d_out, int out_size, void* d_ws, size_t ws_size,
                              hipStream_t stream)
{
  const float* Qin    = (const float*)d_in[0];
  const float* Kin    = (const float*)d_in[1];
  const float* Vin    = (const float*)d_in[2];
  const float* QTrans = (const float*)d_in[3];
  const float* KTrans = (const float*)d_in[4];
  const float* VTrans = (const float*)d_in[5];
  const float* GW     = (const float*)d_in[6];
  const float* Gb     = (const float*)d_in[7];
  const float* outW   = (const float*)d_in[8];
  const float* outb   = (const float*)d_in[9];

  const int B = 2, S = 2048, D = 1024;
  const int M = B * S;                               // 4096
  const long fbytes = (long)M * D * sizeof(float);   // 16 MB
  const long hbytes = (long)M * D * sizeof(short);   // 8 MB

  char* ws = (char*)d_ws;
  float*  Vp  = (float*)(ws);
  float*  Gp  = (float*)(ws + fbytes);
  __bf16* Qh  = (__bf16*)(ws + 2 * fbytes);
  __bf16* Kh  = (__bf16*)(ws + 2 * fbytes + hbytes);
  __bf16* VGh = (__bf16*)(ws + 2 * fbytes + 2 * hbytes);
  float*  ctx = (float*)(ws + 2 * fbytes + 3 * hbytes);   // total 72 MB

  dim3 blk(256);
  dim3 gGemm(D / 64, M / 64);

  // projections
  gemm_bf16_wmma<<<gGemm, blk, 0, stream>>>(Qin, QTrans, nullptr, nullptr, Qh, M, D, D, S, 1);
  gemm_bf16_wmma<<<gGemm, blk, 0, stream>>>(Kin, KTrans, nullptr, nullptr, Kh, M, D, D, S, 1);
  gemm_bf16_wmma<<<gGemm, blk, 0, stream>>>(Vin, VTrans, nullptr, Vp, nullptr, M, D, D, S, 0);
  gemm_bf16_wmma<<<gGemm, blk, 0, stream>>>(Vin, GW, Gb, Gp, nullptr, M, D, D, S, 0);

  const int total = M * D;
  pack_vg<<<dim3(total / 256), blk, 0, stream>>>(Vp, Gp, VGh, S, total);

  attn_fwd<<<dim3(S / 64, HNUM, B), blk, 0, stream>>>(Qh, Kh, VGh, ctx, S);

  // output projection + bias
  gemm_bf16_wmma<<<gGemm, blk, 0, stream>>>(ctx, outW, outb, (float*)d_out, nullptr, M, D, D, S, 0);
}
